// TimeSformer_t_3994319585918
// MI455X (gfx1250) — matrix operand order repaired and checked by reading, not compiled, not run on hardware
//
#include <hip/hip_runtime.h>
#include <hip/hip_bf16.h>
#include <math.h>

// ---------------- types ----------------
typedef __bf16 bf16_t;
typedef __attribute__((ext_vector_type(16))) __bf16 bf16x16;
typedef __attribute__((ext_vector_type(8)))  float  floatx8;

// ---------------- model constants ----------------
// B=2, F=8, DIM=256, H=64, W=64, P=4, HEADS=8, DH=64, DEPTH=3, INNER=512
// hp=wp=16, n_t = 1024 (time seq groups), Ntok_t = 8193, Ntok_s = 2049

// Branchless guarded staging of 8 bf16 elements of a row-major matrix into LDS.
// Two-phase (all loads first, then selects) so loads stay in flight together.
__device__ __forceinline__ void stage_row8(
    const bf16_t* __restrict__ G, bf16_t* dst,
    int grow, int rows, int gk, int K, bool kv8)
{
    if (grow < rows && kv8 && (gk + 8) <= K) {
        *reinterpret_cast<uint4*>(dst) =
            *reinterpret_cast<const uint4*>(G + (long long)grow * K + gk);
    } else {
        const long long rowoff = (long long)(grow < rows ? grow : 0) * K;
        bf16_t v[8];
        bool ok[8];
#pragma unroll
        for (int j = 0; j < 8; ++j) {
            const int kk = gk + j;
            ok[j] = (grow < rows) && (kk < K);
            v[j]  = G[rowoff + (ok[j] ? kk : 0)];   // clamped, always safe
        }
#pragma unroll
        for (int j = 0; j < 8; ++j)
            dst[j] = ok[j] ? v[j] : (bf16_t)0.0f;
    }
}

// Stage 8 columns of a K x N row-major B into N-major LDS tile (transpose).
__device__ __forceinline__ void stage_colmajor8(
    const bf16_t* __restrict__ B, bf16_t (*Bs)[48],
    int gk, int K, int gn0, int N, int bkl, int bnl)
{
    const bool okk = gk < K;
    if (okk && ((N & 7) == 0) && (gn0 + 8) <= N) {
        union { uint4 u; bf16_t h[8]; } tmp;
        tmp.u = *reinterpret_cast<const uint4*>(B + (long long)gk * N + gn0);
#pragma unroll
        for (int j = 0; j < 8; ++j) Bs[bnl + j][bkl] = tmp.h[j];
    } else {
        const long long rowoff = (long long)(okk ? gk : 0) * N;
        bf16_t v[8];
        bool ok[8];
#pragma unroll
        for (int j = 0; j < 8; ++j) {
            const int gn = gn0 + j;
            ok[j] = okk && (gn < N);
            v[j]  = B[rowoff + (ok[j] ? gn : 0)];
        }
#pragma unroll
        for (int j = 0; j < 8; ++j)
            Bs[bnl + j][bkl] = ok[j] ? v[j] : (bf16_t)0.0f;
    }
}

// =====================================================================
// Generic batched bf16 WMMA GEMM:  C = A @ B (+bias[n]) (+residual)
//   A: [M,K] row-major bf16 (batch stride sA elems)
//   B: flagBT==0 : [K,N] row-major ; flagBT==1 : [N,K] row-major
//   C: [M,N] row-major f32 (batch stride sC); residual indexed like C
// Block: 256 threads = 8 waves; block tile 64x128, K step 32.
// Wave (wr,wc) in 2x4 grid computes a 32x32 sub-tile: 2 A-frags x 2 B-frags
// = 4 v_wmma_f32_16x16x32_bf16 per K step.
// Interior tiles stream global->LDS with GLOBAL_LOAD_ASYNC_TO_LDS_B128
// (ASYNCcnt), edge tiles use branchless guarded staging.
// =====================================================================
__global__ __launch_bounds__(256) void gemm_bf16_kernel(
    const bf16_t* __restrict__ A, const bf16_t* __restrict__ B,
    float* __restrict__ C, const float* __restrict__ bias,
    const float* __restrict__ res,
    int M, int N, int K,
    long long sA, long long sB, long long sC, int flagBT)
{
    __shared__ bf16_t As[64][48];    // row stride 96B -> 16B-aligned chunks
    __shared__ bf16_t Bs[128][48];   // stored N-major (Bs[n][k])

    const int tid  = threadIdx.x;
    const int lane = tid & 31;
    const int wave = tid >> 5;
    const int wr   = wave >> 2;       // 0..1
    const int wc   = wave & 3;        // 0..3
    const int rowBase = blockIdx.y * 64;
    const int colBase = blockIdx.x * 128;
    const long long bz = blockIdx.z;

    A += bz * sA;
    B += bz * sB;
    C += bz * sC;
    const float* resp = res ? (res + bz * sC) : nullptr;

    floatx8 acc[2][2] = {};

    const bool kv8   = ((K & 7) == 0);
    const bool kfull = ((K & 31) == 0);
    // Block-uniform fast-path predicates (uniform => scalar branch, EXEC stays all-1s)
    const bool asyncA = kfull && (rowBase + 64 <= M);
    const bool asyncB = (flagBT != 0) && kfull && (colBase + 128 <= N);

    const int nK  = (K + 31) >> 5;
    const int ar  = tid >> 2;          // 0..63  (A tile row)
    const int ak  = (tid & 3) << 3;    // 0,8,16,24
    const int br  = tid >> 1;          // 0..127 (Bt tile row)
    const int bk2 = (tid & 1) << 4;    // 0,16
    const int ckl = tid >> 3;          // 0..31  (k for KxN staging)
    const int cnl = (tid & 7) << 4;    // 0..112 (16 cols per thread)

    for (int kt = 0; kt < nK; ++kt) {
        const int k0 = kt << 5;

        // ---- stage A tile (64x32) ----
        if (asyncA) {
            const uint32_t lo = (uint32_t)(size_t)&As[ar][ak];
            const bf16_t* gp = A + (long long)(rowBase + ar) * K + (k0 + ak);
            asm volatile("global_load_async_to_lds_b128 %0, %1, off"
                         :: "v"(lo), "v"(gp) : "memory");
        } else {
            stage_row8(A, &As[ar][ak], rowBase + ar, M, k0 + ak, K, kv8);
        }

        // ---- stage B tile (128x32, N-major in LDS) ----
        if (flagBT) {
            if (asyncB) {
                const uint32_t lo0 = (uint32_t)(size_t)&Bs[br][bk2];
                const uint32_t lo1 = (uint32_t)(size_t)&Bs[br][bk2 + 8];
                const bf16_t* gp = B + (long long)(colBase + br) * K + (k0 + bk2);
                asm volatile("global_load_async_to_lds_b128 %0, %1, off"
                             :: "v"(lo0), "v"(gp) : "memory");
                asm volatile("global_load_async_to_lds_b128 %0, %1, off"
                             :: "v"(lo1), "v"(gp + 8) : "memory");
            } else {
                stage_row8(B, &Bs[br][bk2],     colBase + br, N, k0 + bk2,     K, kv8);
                stage_row8(B, &Bs[br][bk2 + 8], colBase + br, N, k0 + bk2 + 8, K, kv8);
            }
        } else {
            const int gk = k0 + ckl;
            stage_colmajor8(B, Bs, gk, K, colBase + cnl,     N, ckl, cnl);
            stage_colmajor8(B, Bs, gk, K, colBase + cnl + 8, N, ckl, cnl + 8);
        }

        if (asyncA || asyncB)
            asm volatile("s_wait_asynccnt 0x0" ::: "memory");
        __syncthreads();

        // ---- A fragments (16x32): lanes 0-15 -> K {0..7,16..23}, 16-31 -> K {8..15,24..31}
        const int ak0 = (lane < 16) ? 0 : 8;
        bf16x16 afrag[2];
#pragma unroll
        for (int i = 0; i < 2; ++i) {
            const int arow = wr * 32 + i * 16 + (lane & 15);
#pragma unroll
            for (int e = 0; e < 8; ++e) {
                afrag[i][e]     = As[arow][ak0 + e];
                afrag[i][8 + e] = As[arow][ak0 + 16 + e];
            }
        }
        // ---- B fragments (32x16): lanes 0-15 -> K {0..7,16..23}, 16-31 -> K {8..15,24..31}
        const int bk0 = (lane < 16) ? 0 : 8;
        bf16x16 bfrag[2];
#pragma unroll
        for (int j = 0; j < 2; ++j) {
            const int bc = wc * 32 + j * 16 + (lane & 15);
#pragma unroll
            for (int e = 0; e < 8; ++e) {
                bfrag[j][e]     = Bs[bc][bk0 + e];
                bfrag[j][8 + e] = Bs[bc][bk0 + 16 + e];
            }
        }
#pragma unroll
        for (int i = 0; i < 2; ++i)
#pragma unroll
            for (int j = 0; j < 2; ++j)
                acc[i][j] = __builtin_amdgcn_wmma_f32_16x16x32_bf16(
                    false, afrag[i], false, bfrag[j], (short)0, acc[i][j],
                    false, false);
        __syncthreads();
    }

    // ---- store C: VGPR r -> M = r (+8 for lanes 16-31), N = lane&15
    const int lm = (lane >> 4) << 3;
    const int ln = lane & 15;
#pragma unroll
    for (int i = 0; i < 2; ++i) {
        const int m0 = rowBase + wr * 32 + i * 16 + lm;
#pragma unroll
        for (int j = 0; j < 2; ++j) {
            const int n = colBase + wc * 32 + j * 16 + ln;
            if (n >= N) continue;
#pragma unroll
            for (int r = 0; r < 8; ++r) {
                const int m = m0 + r;
                if (m >= M) continue;
                float v = acc[i][j][r];
                if (bias) v += bias[n];
                const long long idx = (long long)m * N + n;
                if (resp) v += resp[idx];
                C[idx] = v;
            }
        }
    }
}

// =====================================================================
// Elementwise / reduction kernels (all wave32-aware, 8 rows per block)
// =====================================================================
__global__ void cvt_f32_bf16_kernel(const float* __restrict__ x,
                                    bf16_t* __restrict__ y, long long n)
{
    long long i = (long long)blockIdx.x * 256 + threadIdx.x;
    if (i < n) y[i] = (bf16_t)x[i];
}

__global__ void layernorm_kernel(const float* __restrict__ X,
                                 const float* __restrict__ g,
                                 const float* __restrict__ bta,
                                 bf16_t* __restrict__ Y, int rows)
{
    const int lane = threadIdx.x & 31;
    const int row  = blockIdx.x * 8 + (threadIdx.x >> 5);
    if (row >= rows) return;
    const float* x = X + (long long)row * 256;
    float s = 0.f, s2 = 0.f;
#pragma unroll
    for (int i = lane; i < 256; i += 32) { float v = x[i]; s += v; s2 += v * v; }
#pragma unroll
    for (int o = 16; o > 0; o >>= 1) {
        s  += __shfl_xor(s,  o, 32);
        s2 += __shfl_xor(s2, o, 32);
    }
    const float mean = s * (1.f / 256.f);
    const float var  = s2 * (1.f / 256.f) - mean * mean;
    const float inv  = rsqrtf(var + 1e-5f);
    bf16_t* y = Y + (long long)row * 256;
#pragma unroll
    for (int i = lane; i < 256; i += 32)
        y[i] = (bf16_t)((x[i] - mean) * inv * g[i] + bta[i]);
}

__global__ void softmax_kernel(const float* __restrict__ S,
                               bf16_t* __restrict__ P, int rows, int L)
{
    const int lane = threadIdx.x & 31;
    const int row  = blockIdx.x * 8 + (threadIdx.x >> 5);
    if (row >= rows) return;
    const float* s = S + (long long)row * L;
    bf16_t* p = P + (long long)row * L;
    float mx = -3.4e38f;
    for (int i = lane; i < L; i += 32) mx = fmaxf(mx, s[i]);
#pragma unroll
    for (int o = 16; o > 0; o >>= 1) mx = fmaxf(mx, __shfl_xor(mx, o, 32));
    float sum = 0.f;
    for (int i = lane; i < L; i += 32) sum += __expf(s[i] - mx);
#pragma unroll
    for (int o = 16; o > 0; o >>= 1) sum += __shfl_xor(sum, o, 32);
    const float inv = 1.f / sum;
    for (int i = lane; i < L; i += 32) p[i] = (bf16_t)(__expf(s[i] - mx) * inv);
}

// sin/cos tables: frame (8x64) + image (256x64), matching reference formulas
__global__ void sincos_kernel(float* __restrict__ fsin, float* __restrict__ fcos,
                              float* __restrict__ isin, float* __restrict__ icos)
{
    const int id = blockIdx.x * 256 + threadIdx.x;
    if (id < 512) {
        const int fi = id >> 6, d = id & 63;
        const int i = d & 31;
        const float inv = __expf(-((2.0f * i) / 64.0f) * 9.2103403720f); // ln(1e4)
        const float a = (float)fi * inv;
        fsin[id] = __sinf(a);
        fcos[id] = __cosf(a);
    }
    if (id < 16384) {
        const int s = id >> 6, d = id & 63;
        const int hp = s >> 4, wq = s & 15;
        const int base = d >> 1;
        float a;
        if (base < 16) {
            const float scale = exp2f(base * (2.3219280949f / 15.0f)); // log2(5)
            a = (-1.0f + 2.0f * hp / 15.0f) * scale * 3.14159265358979f;
        } else {
            const int ss = base - 16;
            const float scale = exp2f(ss * (2.3219280949f / 15.0f));
            a = (-1.0f + 2.0f * wq / 15.0f) * scale * 3.14159265358979f;
        }
        isin[id] = __sinf(a);
        icos[id] = __cosf(a);
    }
}

// global-token attention inputs: Qw[BH,64] (scaled, no rope), Kw/Vw[BH,Ntok,64]
__global__ void build_wgt_kernel(const float* __restrict__ qkv,
                                 bf16_t* __restrict__ Qw, bf16_t* __restrict__ Kw,
                                 bf16_t* __restrict__ Vw, int Ntok)
{
    long long gid = (long long)blockIdx.x * 256 + threadIdx.x;
    const long long total = 16LL * Ntok * 32;
    if (gid >= total) return;
    const int j = gid & 31;
    long long t = gid >> 5;
    const int tok = (int)(t % Ntok);
    const int bh  = (int)(t / Ntok);
    const int b = bh >> 3, h = bh & 7;
    const float* base = qkv + ((long long)b * Ntok + tok) * 1536 + h * 64;
    const long long o = ((long long)bh * Ntok + tok) * 64 + 2 * j;
    Kw[o]     = (bf16_t)base[512 + 2 * j];
    Kw[o + 1] = (bf16_t)base[512 + 2 * j + 1];
    Vw[o]     = (bf16_t)base[1024 + 2 * j];
    Vw[o + 1] = (bf16_t)base[1024 + 2 * j + 1];
    if (tok == 0) {
        Qw[bh * 64 + 2 * j]     = (bf16_t)(base[2 * j] * 0.125f);
        Qw[bh * 64 + 2 * j + 1] = (bf16_t)(base[2 * j + 1] * 0.125f);
    }
}

// per-sequence Q/K/V with rotary; Kr/Vr rows 1..Lq, row 0 filled separately
__global__ void build_qkv_kernel(const float* __restrict__ qkv,
                                 bf16_t* __restrict__ Qr, bf16_t* __restrict__ Kr,
                                 bf16_t* __restrict__ Vr,
                                 const float* __restrict__ sinT,
                                 const float* __restrict__ cosT,
                                 int Ntok, int seqr, int Lq, int timeMode)
{
    long long gid = (long long)blockIdx.x * 256 + threadIdx.x;
    const long long total = 16LL * seqr * Lq * 32;
    if (gid >= total) return;
    const int j = gid & 31;
    long long t = gid >> 5;
    const int row   = (int)(t % Lq);
    const int batch = (int)(t / Lq);
    int bh, tok;
    if (timeMode) { const int si = batch % seqr; bh = batch / seqr; tok = 1 + row * seqr + si; }
    else          { const int fi = batch % seqr; bh = batch / seqr; tok = 1 + fi * Lq + row; }
    const int b = bh >> 3, h = bh & 7;
    const float* base = qkv + ((long long)b * Ntok + tok) * 1536 + h * 64;
    const float q0 = base[2 * j],        q1 = base[2 * j + 1];
    const float k0 = base[512 + 2 * j],  k1 = base[512 + 2 * j + 1];
    const float v0 = base[1024 + 2 * j], v1 = base[1024 + 2 * j + 1];
    const float* sn = sinT + row * 64;
    const float* cn = cosT + row * 64;
    const float s0 = sn[2 * j], s1 = sn[2 * j + 1];
    const float c0 = cn[2 * j], c1 = cn[2 * j + 1];
    // rotate_every_two: out[2j] = x[2j]*c - x[2j+1]*s ; out[2j+1] = x[2j+1]*c + x[2j]*s
    const float qo0 = q0 * c0 - q1 * s0;
    const float qo1 = q1 * c1 + q0 * s1;
    const float ko0 = k0 * c0 - k1 * s0;
    const float ko1 = k1 * c1 + k0 * s1;
    const long long qo = ((long long)batch * Lq + row) * 64 + 2 * j;
    Qr[qo]     = (bf16_t)(qo0 * 0.125f);
    Qr[qo + 1] = (bf16_t)(qo1 * 0.125f);
    const int Lk = Lq + 1;
    const long long ko = ((long long)batch * Lk + row + 1) * 64 + 2 * j;
    Kr[ko]     = (bf16_t)ko0;
    Kr[ko + 1] = (bf16_t)ko1;
    Vr[ko]     = (bf16_t)v0;
    Vr[ko + 1] = (bf16_t)v1;
}

__global__ void fill_seq_wgt_kernel(const float* __restrict__ qkv,
                                    bf16_t* __restrict__ Kr, bf16_t* __restrict__ Vr,
                                    int Ntok, int seqr, int Lq)
{
    long long gid = (long long)blockIdx.x * 256 + threadIdx.x;
    const long long total = 16LL * seqr * 32;
    if (gid >= total) return;
    const int j = gid & 31;
    const int batch = (int)(gid >> 5);
    const int bh = batch / seqr;
    const int b = bh >> 3, h = bh & 7;
    const float* base = qkv + (long long)b * Ntok * 1536 + h * 64;  // token 0
    const int Lk = Lq + 1;
    const long long o = (long long)batch * Lk * 64 + 2 * j;
    Kr[o]     = (bf16_t)base[512 + 2 * j];
    Kr[o + 1] = (bf16_t)base[512 + 2 * j + 1];
    Vr[o]     = (bf16_t)base[1024 + 2 * j];
    Vr[o + 1] = (bf16_t)base[1024 + 2 * j + 1];
}

__global__ void merge_attn_kernel(const float* __restrict__ wgt_out,
                                  const float* __restrict__ attn_out,
                                  bf16_t* __restrict__ cat,
                                  int Ntok, int seqr, int Lq, int timeMode)
{
    long long gid = (long long)blockIdx.x * 256 + threadIdx.x;
    const long long total = 2LL * Ntok * 512;
    if (gid >= total) return;
    const int ch = gid & 511;
    long long t = gid >> 9;
    const int tok = (int)(t % Ntok);
    const int b   = (int)(t / Ntok);
    const int h = ch >> 6, d = ch & 63;
    const int bh = b * 8 + h;
    float v;
    if (tok == 0) {
        v = wgt_out[bh * 64 + d];
    } else {
        const int idx = tok - 1;
        const int n = timeMode ? seqr : Lq;
        const int fi = idx / n, si = idx % n;
        long long o;
        if (timeMode) o = (((long long)bh * seqr + si) * Lq + fi) * 64 + d;
        else          o = (((long long)bh * seqr + fi) * Lq + si) * 64 + d;
        v = attn_out[o];
    }
    cat[gid] = (bf16_t)v;
}

__global__ void glu_kernel(const float* __restrict__ h,
                           bf16_t* __restrict__ u, long long rows)
{
    long long gid = (long long)blockIdx.x * 256 + threadIdx.x;
    const long long total = rows * 1024;
    if (gid >= total) return;
    const long long row = gid >> 10;
    const int jc = gid & 1023;
    const float a = h[row * 2048 + jc];
    const float g = h[row * 2048 + 1024 + jc];
    const float gl = 0.5f * g * (1.0f + erff(g * 0.70710678f));
    u[gid] = (bf16_t)(a * gl);
}

// (b,f,c,h,w) -> bf16 tokens (b, f*h*wp, p*c) per reference patchify
__global__ void embed_gather_kernel(const float* __restrict__ fr,
                                    bf16_t* __restrict__ t)
{
    long long gid = (long long)blockIdx.x * 256 + threadIdx.x;
    const long long total = 2LL * 8192 * 1024;
    if (gid >= total) return;
    const int ch = gid & 1023;
    long long tk = gid >> 10;
    const int tok = (int)(tk % 8192);
    const int b   = (int)(tk / 8192);
    const int p_i = ch >> 8, c_i = ch & 255;
    const int wp_i = tok & 15, h_i = (tok >> 4) & 63, f_i = tok >> 10;
    const int w_i = wp_i * 4 + p_i;
    const long long src =
        ((((long long)b * 8 + f_i) * 256 + c_i) * 64 + h_i) * 64 + w_i;
    t[gid] = (bf16_t)fr[src];
}

__global__ void set_dy_kernel(float* __restrict__ xt, const float* __restrict__ dy)
{
    xt[(long long)blockIdx.x * 8193 * 256 + threadIdx.x] = dy[threadIdx.x];
}

// time->spatial token remap (+ wgt token appended at end)
__global__ void scatter_ts_kernel(const float* __restrict__ xs,
                                  const float* __restrict__ xt,
                                  float* __restrict__ xsp)
{
    long long gid = (long long)blockIdx.x * 256 + threadIdx.x;
    const long long total = 2LL * 2049 * 256;
    if (gid >= total) return;
    const int ch = gid & 255;
    long long t = gid >> 8;
    const int s = (int)(t % 2049);
    const int b = (int)(t / 2049);
    float v;
    if (s == 2048) {
        v = xt[(long long)b * 8193 * 256 + ch];          // wgt = x[:, 0]
    } else {
        const int f_i = s >> 8, rem = s & 255, hp = rem >> 4, wq = rem & 15;
        const int pp = ch >> 4, c2 = ch & 15, ph = pp >> 2, pw = pp & 3;
        const int tok = (f_i * 64 + hp * 4 + ph) * 16 + wq;
        const int ich = pw * 16 + c2;
        v = xs[((long long)b * 8192 + tok) * 64 + ich];
    }
    xsp[gid] = v;
}

// spatial->time token remap (+ wgt token appended at end)
__global__ void scatter_st_kernel(const float* __restrict__ xtp,
                                  const float* __restrict__ xsp,
                                  float* __restrict__ xt)
{
    long long gid = (long long)blockIdx.x * 256 + threadIdx.x;
    const long long total = 2LL * 8193 * 256;
    if (gid >= total) return;
    const int ch = gid & 255;
    long long tt = gid >> 8;
    const int t = (int)(tt % 8193);
    const int b = (int)(tt / 8193);
    float v;
    if (t == 8192) {
        v = xsp[(long long)b * 2049 * 256 + ch];         // wgt = x[:, 0]
    } else {
        const int wp = t & 15, h = (t >> 4) & 63, f_i = t >> 10;
        const int p_i = ch >> 6, c3 = ch & 63;
        const int w = wp * 4 + p_i;
        const int hp = h >> 2;
        const int wq = (h & 3) * 4 + (w >> 4);
        const int ph = (w >> 2) & 3, pw = w & 3;
        const int tok = f_i * 256 + hp * 16 + wq;
        const int ich = (ph * 4 + pw) * 64 + c3;
        v = xtp[((long long)b * 2048 + tok) * 1024 + ich];
    }
    xt[gid] = v;
}

__global__ void copy_out_kernel(const float* __restrict__ xsp, float* __restrict__ out)
{
    out[blockIdx.x * 256 + threadIdx.x] =
        xsp[(long long)blockIdx.x * 2049 * 256 + threadIdx.x];
}

// =====================================================================
// Host orchestration
// =====================================================================
extern "C" void kernel_launch(void* const* d_in, const int* in_sizes, int n_in,
                              void* d_out, int out_size, void* d_ws, size_t ws_size,
                              hipStream_t stream)
{
    (void)in_sizes; (void)n_in; (void)out_size; (void)ws_size;

    const float* frame = (const float*)d_in[0];
    const float* W_emb = (const float*)d_in[1];
    const float* b_emb = (const float*)d_in[2];
    const float* W_ts  = (const float*)d_in[3];
    const float* b_ts  = (const float*)d_in[4];
    const float* W_st  = (const float*)d_in[5];
    const float* b_st  = (const float*)d_in[6];
    const float* dyw   = (const float*)d_in[7];

    // ---- bump allocator over d_ws ----
    char* wsb = (char*)d_ws;
    size_t off = 0;
    auto alloc = [&](size_t bytes) -> char* {
        char* p = wsb + off;
        off = (off + bytes + 255) & ~(size_t)255;
        return p;
    };

    float* fsin = (float*)alloc(512 * 4);
    float* fcos = (float*)alloc(512 * 4);
    float* isin = (float*)alloc(16384 * 4);
    float* icos = (float*)alloc(16384 * 4);

    // bf16 weight arena
    bf16_t* wemb_bf = (bf16_t*)alloc(262144 * 2);
    bf16_t* wts_bf  = (bf16_t*)alloc(16384 * 2);
    bf16_t* wst_bf  = (bf16_t*)alloc(262144 * 2);
    bf16_t *tWqkv[3], *tWout[3], *sWqkv[3], *sWout[3], *W1bf[3], *W2bf[3];
    for (int l = 0; l < 3; ++l) {
        tWqkv[l] = (bf16_t*)alloc(393216 * 2);
        tWout[l] = (bf16_t*)alloc(131072 * 2);
        sWqkv[l] = (bf16_t*)alloc(393216 * 2);
        sWout[l] = (bf16_t*)alloc(131072 * 2);
        W1bf[l]  = (bf16_t*)alloc(524288 * 2);
        W2bf[l]  = (bf16_t*)alloc(262144 * 2);
    }

    float*  x_t   = (float*)alloc(4194816ULL * 4);   // (2,8193,256)
    float*  xsp   = (float*)alloc(1049088ULL * 4);   // (2,2049,256)
    bf16_t* x_bf  = (bf16_t*)alloc(4194816ULL * 2);
    bf16_t* ln_bf = (bf16_t*)alloc(4194816ULL * 2);
    float*  qkv   = (float*)alloc(25168896ULL * 4);  // (2,8193,1536); also emb staging
    bf16_t* Qw    = (bf16_t*)alloc(1024 * 2);
    float*  wgtO  = (float*)alloc(1024 * 4);
    bf16_t* Kw    = (bf16_t*)alloc(8389632ULL * 2);  // 16*8193*64
    bf16_t* Vw    = (bf16_t*)alloc(8389632ULL * 2);
    bf16_t* Qr    = (bf16_t*)alloc(8388608ULL * 2);  // 16384*8*64
    bf16_t* Kr    = (bf16_t*)alloc(9437184ULL * 2);  // 16384*9*64
    bf16_t* Vr    = (bf16_t*)alloc(9437184ULL * 2);
    float*  simb  = (float*)alloc(8421376ULL * 4);   // 128*256*257
    bf16_t* probs = (bf16_t*)alloc(8421376ULL * 2);
    float*  aout  = (float*)alloc(8388608ULL * 4);   // 16384*8*64
    bf16_t* acat  = (bf16_t*)alloc(8389632ULL * 2);  // 16386*512
    float*  gtmp  = (float*)alloc(8392704ULL * 4);   // 4098*2048
    bf16_t* glu_b = (bf16_t*)alloc(4196352ULL * 2);  // 4098*1024
    bf16_t* emb_bf = (bf16_t*)qkv;                   // 16.8M bf16 fits in qkv region

    auto gblk = [](long long n) { return (unsigned)((n + 255) / 256); };
    auto cvt = [&](const float* s, bf16_t* d, long long n) {
        cvt_f32_bf16_kernel<<<gblk(n), 256, 0, stream>>>(s, d, n);
    };
    auto gemm = [&](const bf16_t* A, const bf16_t* B, float* C,
                    const float* bias, const float* res,
                    int M, int N, int K, int batch,
                    long long sA, long long sB, long long sC, int flagBT) {
        dim3 g((N + 127) / 128, (M + 63) / 64, batch);
        gemm_bf16_kernel<<<g, 256, 0, stream>>>(A, B, C, bias, res,
                                                M, N, K, sA, sB, sC, flagBT);
    };

    // ---- tables + weights to bf16 ----
    sincos_kernel<<<64, 256, 0, stream>>>(fsin, fcos, isin, icos);
    cvt(W_emb, wemb_bf, 262144);
    cvt(W_ts,  wts_bf,  16384);
    cvt(W_st,  wst_bf,  262144);
    for (int l = 0; l < 3; ++l) {
        const int base = 8 + l * 16;
        cvt((const float*)d_in[base + 2],  tWqkv[l], 393216);
        cvt((const float*)d_in[base + 3],  tWout[l], 131072);
        cvt((const float*)d_in[base + 7],  sWqkv[l], 393216);
        cvt((const float*)d_in[base + 8],  sWout[l], 131072);
        cvt((const float*)d_in[base + 12], W1bf[l],  524288);
        cvt((const float*)d_in[base + 14], W2bf[l],  262144);
    }

    // ---- patch embedding ----
    embed_gather_kernel<<<gblk(2LL * 8192 * 1024), 256, 0, stream>>>(frame, emb_bf);
    gemm(emb_bf, wemb_bf, x_t + 256, b_emb, nullptr,
         8192, 256, 1024, 2, 8192LL * 1024, 0, 8193LL * 256, 0);
    set_dy_kernel<<<2, 256, 0, stream>>>(x_t, dyw);

    // ---- attention pass (in-place residual on x) ----
    auto run_attention = [&](float* x, int Ntok, int seqr, int Lq, int timeMode,
                             const float* sinT, const float* cosT,
                             const bf16_t* Wqkv_bf, const bf16_t* Wout_bf,
                             const float* ln_g, const float* ln_b,
                             const float* bout) {
        const int rows = 2 * Ntok;
        layernorm_kernel<<<(rows + 7) / 8, 256, 0, stream>>>(x, ln_g, ln_b, ln_bf, rows);
        gemm(ln_bf, Wqkv_bf, qkv, nullptr, nullptr, rows, 1536, 256, 1, 0, 0, 0, 0);

        // global-token attention over the full sequence
        build_wgt_kernel<<<gblk(16LL * Ntok * 32), 256, 0, stream>>>(qkv, Qw, Kw, Vw, Ntok);
        gemm(Qw, Kw, simb, nullptr, nullptr, 1, Ntok, 64, 16,
             64, (long long)Ntok * 64, Ntok, 1);
        softmax_kernel<<<(16 + 7) / 8, 256, 0, stream>>>(simb, probs, 16, Ntok);
        gemm(probs, Vw, wgtO, nullptr, nullptr, 1, 64, Ntok, 16,
             Ntok, (long long)Ntok * 64, 64, 0);

        // per-sequence attention (wgt token prepended to K/V)
        const int batches = 16 * seqr;
        const int Lk = Lq + 1;
        build_qkv_kernel<<<gblk(16LL * seqr * Lq * 32), 256, 0, stream>>>(
            qkv, Qr, Kr, Vr, sinT, cosT, Ntok, seqr, Lq, timeMode);
        fill_seq_wgt_kernel<<<gblk(16LL * seqr * 32), 256, 0, stream>>>(
            qkv, Kr, Vr, Ntok, seqr, Lq);
        gemm(Qr, Kr, simb, nullptr, nullptr, Lq, Lk, 64, batches,
             (long long)Lq * 64, (long long)Lk * 64, (long long)Lq * Lk, 1);
        softmax_kernel<<<((long long)batches * Lq + 7) / 8, 256, 0, stream>>>(
            simb, probs, batches * Lq, Lk);
        gemm(probs, Vr, aout, nullptr, nullptr, Lq, 64, Lk, batches,
             (long long)Lq * Lk, (long long)Lk * 64, (long long)Lq * 64, 0);

        merge_attn_kernel<<<gblk(2LL * Ntok * 512), 256, 0, stream>>>(
            wgtO, aout, acat, Ntok, seqr, Lq, timeMode);
        // output projection + bias + residual (in place)
        gemm(acat, Wout_bf, x, bout, x, rows, 256, 512, 1, 0, 0, 0, 0);
    };

    auto run_ff = [&](float* x, const bf16_t* w1, const bf16_t* w2,
                      const float* ln_g, const float* ln_b,
                      const float* b1, const float* b2) {
        const int rows = 2 * 2049;
        layernorm_kernel<<<(rows + 7) / 8, 256, 0, stream>>>(x, ln_g, ln_b, ln_bf, rows);
        gemm(ln_bf, w1, gtmp, b1, nullptr, rows, 2048, 256, 1, 0, 0, 0, 0);
        glu_kernel<<<gblk((long long)rows * 1024), 256, 0, stream>>>(gtmp, glu_b, rows);
        gemm(glu_b, w2, x, b2, x, rows, 256, 1024, 1, 0, 0, 0, 0);
    };

    // ---- layers ----
    for (int l = 0; l < 3; ++l) {
        const int base = 8 + l * 16;
        // time attention on x_t (8193 tokens; sequences over f=8 frames)
        run_attention(x_t, 8193, 1024, 8, 1, fsin, fcos,
                      tWqkv[l], tWout[l],
                      (const float*)d_in[base + 0], (const float*)d_in[base + 1],
                      (const float*)d_in[base + 4]);
        // time -> spatial projection
        cvt(x_t, x_bf, 4194816);
        gemm(x_bf + 256, wts_bf, gtmp, b_ts, nullptr,
             8192, 64, 256, 2, 8193LL * 256, 0, 8192LL * 64, 0);
        scatter_ts_kernel<<<gblk(2LL * 2049 * 256), 256, 0, stream>>>(gtmp, x_t, xsp);
        // spatial attention on xsp (2049 tokens; sequences over n=256 patches)
        run_attention(xsp, 2049, 8, 256, 0, isin, icos,
                      sWqkv[l], sWout[l],
                      (const float*)d_in[base + 5], (const float*)d_in[base + 6],
                      (const float*)d_in[base + 9]);
        // feed-forward (GeGLU)
        run_ff(xsp, W1bf[l], W2bf[l],
               (const float*)d_in[base + 10], (const float*)d_in[base + 11],
               (const float*)d_in[base + 13], (const float*)d_in[base + 15]);
        // spatial -> time projection
        cvt(xsp, x_bf, 1049088);
        gemm(x_bf + 256, wst_bf, gtmp, b_st, nullptr,
             2048, 1024, 256, 2, 2049LL * 256, 0, 2048LL * 1024, 0);
        scatter_st_kernel<<<gblk(2LL * 8193 * 256), 256, 0, stream>>>(gtmp, xsp, x_t);
    }

    copy_out_kernel<<<2, 256, 0, stream>>>(xsp, (float*)d_out);
}
